// VC_MODEL_37512244363892
// MI455X (gfx1250) — matrix operand order repaired and checked by reading, not compiled, not run on hardware
//
#include <hip/hip_runtime.h>

// ---------------- CDNA5 WMMA types ----------------
typedef __attribute__((ext_vector_type(16))) __bf16 v16bf;
typedef __attribute__((ext_vector_type(8)))  float  v8f;

union FragAB { v16bf v; uint4 q[2]; };
union FragC  { v8f   v; float f[8]; };

__device__ __forceinline__ unsigned short f2bf(float f){
    unsigned int u; __builtin_memcpy(&u, &f, 4);
    unsigned int r = u + 0x7FFFu + ((u >> 16) & 1u);
    return (unsigned short)(r >> 16);
}

#define CDIV(a,b) (((a)+(b)-1)/(b))

// ---------------- im2col (f32 NCH -> bf16 MxKp, padded, optional reflect/flip) --------
__global__ void k_im2col(const float* __restrict__ X, unsigned short* __restrict__ A,
                         int Cin, int ci_off, int Ctot, int Tin, int Tout,
                         int KW, int stride, int pad, int reflect, int flip,
                         int Kp, long long M, int K){
    long long id = (long long)blockIdx.x * blockDim.x + threadIdx.x;
    long long tot = M * (long long)Kp;
    if (id >= tot) return;
    int k = (int)(id % Kp);
    long long mrow = id / Kp;
    float v = 0.f;
    if (k < K){
        int b = (int)(mrow / Tout), t = (int)(mrow % Tout);
        int ci = k / KW, kk = k % KW;
        int tt = flip ? (Tout - 1 - t) : t;
        int tin = tt * stride + kk - pad;
        bool ok = true;
        if (tin < 0)    { if (reflect) tin = -tin;            else ok = false; }
        if (tin >= Tin) { if (reflect) tin = 2*Tin - 2 - tin; else ok = false; }
        if (ok) v = X[((size_t)b * Ctot + ci_off + ci) * Tin + tin];
    }
    A[id] = f2bf(v);
}

// ------- pack weight (N x K row-major f32, row stride ldw) into WMMA B fragments -------
// layout: frag(nt,kt) = 32 lanes x 16 bf16 contiguous;  lane: n = nt*16+lane%16,
//         k = kt*32 + (lane<16?0:8) + (e<8 ? e : e+8)   (e = 0..15)
__global__ void k_pack(const float* __restrict__ W, unsigned short* __restrict__ P,
                       int N, int K, int ldw, int KT, int NT){
    long long id = (long long)blockIdx.x * blockDim.x + threadIdx.x;
    int lane = (int)(id & 31);
    long long f = id >> 5;
    if (f >= (long long)NT * KT) return;
    int nt = (int)(f / KT), kt = (int)(f % KT);
    int n  = nt * 16 + (lane & 15);
    int kb = kt * 32 + ((lane & 16) ? 8 : 0);
    unsigned short* dst = P + ((size_t)f * 32 + lane) * 16;
    #pragma unroll
    for (int e = 0; e < 16; ++e){
        int k = kb + (e < 8 ? e : e + 8);
        float v = (n < N && k < K) ? W[(size_t)n * ldw + k] : 0.f;
        dst[e] = f2bf(v);
    }
}

// ---------------- generic bf16 WMMA GEMM: out = act(A @ B + bias) ----------------
// A: bf16 M x Kp row-major (M multiple of 16, Kp multiple of 32)
// Bp: packed fragments.  mode bits: 1=leaky, 2=store NCH (m=b*T+t), 4=accumulate
__global__ void k_gemm(const unsigned short* __restrict__ A,
                       const unsigned short* __restrict__ Bp,
                       const float* __restrict__ bias, float* __restrict__ out,
                       int M, int N, int Kp, int NT, int tiles, int mode,
                       int T, int Cout, int n_off){
    int wave = threadIdx.x >> 5;
    int tile = blockIdx.x * 8 + wave;
    if (tile >= tiles) return;
    int lane = threadIdx.x & 31;
    int mt = tile / NT, nt = tile % NT;
    int KT = Kp >> 5;
    FragAB fa, fb; FragC acc;
    #pragma unroll
    for (int r = 0; r < 8; ++r) acc.f[r] = 0.f;
    int abase = (lane & 16) ? 8 : 0;
    const unsigned short* Arow  = A + (size_t)(mt * 16 + (lane & 15)) * Kp;
    const unsigned short* Bbase = Bp + (size_t)nt * KT * 512;
    for (int kt = 0; kt < KT; ++kt){
        __builtin_prefetch(Bbase + (size_t)(kt + 1) * 512 + lane * 16, 0, 0);
        fa.q[0] = *reinterpret_cast<const uint4*>(Arow + kt * 32 + abase);
        fa.q[1] = *reinterpret_cast<const uint4*>(Arow + kt * 32 + 16 + abase);
        const uint4* pb = reinterpret_cast<const uint4*>(Bbase + (size_t)kt * 512 + lane * 16);
        fb.q[0] = pb[0]; fb.q[1] = pb[1];
        acc.v = __builtin_amdgcn_wmma_f32_16x16x32_bf16(false, fa.v, false, fb.v,
                                                        (short)0, acc.v, false, false);
    }
    int mofs = (lane & 16) ? 8 : 0;
    int n = nt * 16 + (lane & 15);
    if (n >= N) return;
    float bv = bias[n];
    #pragma unroll
    for (int r = 0; r < 8; ++r){
        int m = mt * 16 + r + mofs;
        if (m >= M) continue;
        float v = acc.f[r] + bv;
        if (mode & 1) v = v < 0.f ? 0.01f * v : v;
        size_t idx;
        if (mode & 2){ int b = m / T, t = m % T; idx = ((size_t)b * Cout + n + n_off) * T + t; }
        else          idx = (size_t)m * N + n;
        if (mode & 4) out[idx] += v; else out[idx] = v;
    }
}

// ---------------- persistent bidirectional-GRU direction kernel ----------------
// 1 workgroup, 16 waves.  h kept in LDS (f32 master + bf16 WMMA copy).
// wave w handles column groups {w, w+16}; for each group it owns the r/z/n tiles
// (nt = gate*32+cg) so the gate math needs no cross-wave gh exchange.
__global__ void __launch_bounds__(512) k_gru(const float* __restrict__ xg,
                                             const unsigned short* __restrict__ Whh,
                                             const float* __restrict__ bhh,
                                             const float* __restrict__ h0,
                                             float* __restrict__ out, int T, int reverse){
    __shared__ unsigned short hbf[16 * 512];
    __shared__ float          hf [16 * 512];
    int tid = threadIdx.x;
    for (int i = tid; i < 16 * 512; i += 512){
        int r = i >> 9, c = i & 511;
        float v = (r < 8) ? h0[(size_t)r * 512 + c] : 0.f;
        hf[i] = v; hbf[i] = f2bf(v);
    }
    __syncthreads();
    int wave = tid >> 5, lane = tid & 31;
    int abase = (lane & 16) ? 8 : 0;
    int mofs  = (lane & 16) ? 8 : 0;
    int nloc  = lane & 15;
    for (int t = 0; t < T; ++t){
        FragC acc[2][3];
        for (int g = 0; g < 2; ++g){
            int cg = wave + 16 * g;
            #pragma unroll
            for (int q = 0; q < 3; ++q)
                #pragma unroll
                for (int r = 0; r < 8; ++r) acc[g][q].f[r] = 0.f;
            const unsigned short* hrow = hbf + (size_t)(lane & 15) * 512;
            for (int kt = 0; kt < 16; ++kt){
                FragAB fa;
                fa.q[0] = *reinterpret_cast<const uint4*>(hrow + kt * 32 + abase);
                fa.q[1] = *reinterpret_cast<const uint4*>(hrow + kt * 32 + 16 + abase);
                #pragma unroll
                for (int gate = 0; gate < 3; ++gate){
                    int nt = gate * 32 + cg;
                    FragAB fb;
                    const uint4* pb = reinterpret_cast<const uint4*>(
                        Whh + ((size_t)(nt * 16 + kt) * 32 + lane) * 16);
                    fb.q[0] = pb[0]; fb.q[1] = pb[1];
                    acc[g][gate].v = __builtin_amdgcn_wmma_f32_16x16x32_bf16(
                        false, fa.v, false, fb.v, (short)0, acc[g][gate].v, false, false);
                }
            }
        }
        __syncthreads();   // all reads of hbf complete before updates
        for (int g = 0; g < 2; ++g){
            int cg = wave + 16 * g;
            int c  = cg * 16 + nloc;
            float br = bhh[c], bz = bhh[512 + c], bn = bhh[1024 + c];
            #pragma unroll
            for (int r = 0; r < 8; ++r){
                int m = r + mofs;
                float xr = 0.f, xz = 0.f, xn = 0.f;
                if (m < 8){
                    const float* row = xg + ((size_t)m * T + t) * 1536;
                    xr = row[c]; xz = row[512 + c]; xn = row[1024 + c];
                }
                float hr = acc[g][0].f[r] + br;
                float hz = acc[g][1].f[r] + bz;
                float hn = acc[g][2].f[r] + bn;
                float rg = 1.f / (1.f + __expf(-(xr + hr)));
                float zg = 1.f / (1.f + __expf(-(xz + hz)));
                float ng = tanhf(xn + rg * hn);
                float hold = hf[(size_t)m * 512 + c];
                float hnew = (1.f - zg) * ng + zg * hold;
                hf [(size_t)m * 512 + c] = hnew;
                hbf[(size_t)m * 512 + c] = f2bf(hnew);
                if (m < 8){
                    int ts = reverse ? (T - 1 - t) : t;
                    out[((size_t)m * T + ts) * 512 + c] = hnew;
                }
            }
        }
        __syncthreads();
    }
}

// ---------------- group norm over (B,C,T), one block per (b, group) ----------------
__global__ void k_groupnorm(const float* __restrict__ X, float* __restrict__ Y,
                            const float* __restrict__ w, const float* __restrict__ bias,
                            int C, int T, int groups, int act){
    int b = blockIdx.x / groups, g = blockIdx.x % groups;
    int cg = C / groups;
    long long n = (long long)cg * T;
    const float* base = X + ((size_t)b * C + (size_t)g * cg) * T;
    __shared__ float s1[256], s2[256];
    float sum = 0.f, sq = 0.f;
    for (long long i = threadIdx.x; i < n; i += blockDim.x){
        float v = base[i]; sum += v; sq += v * v;
    }
    s1[threadIdx.x] = sum; s2[threadIdx.x] = sq; __syncthreads();
    for (int st = 128; st > 0; st >>= 1){
        if ((int)threadIdx.x < st){ s1[threadIdx.x] += s1[threadIdx.x + st];
                                    s2[threadIdx.x] += s2[threadIdx.x + st]; }
        __syncthreads();
    }
    float mu  = s1[0] / (float)n;
    float var = s2[0] / (float)n - mu * mu;
    float inv = rsqrtf(var + 1e-5f);
    float* outp = Y + ((size_t)b * C + (size_t)g * cg) * T;
    for (long long i = threadIdx.x; i < n; i += blockDim.x){
        int c = g * cg + (int)(i / T);
        float v = (base[i] - mu) * inv * w[c] + bias[c];
        if (act) v = v < 0.f ? 0.01f * v : v;
        outp[i] = v;
    }
}

// ---------------- per-(b,t) channel L2 normalize in place ----------------
__global__ void k_l2norm(float* __restrict__ X, int C, int T){
    long long i = (long long)blockIdx.x * blockDim.x + threadIdx.x;
    long long tot = (long long)8 * T;
    if (i >= tot) return;
    int b = (int)(i / T), t = (int)(i % T);
    float s = 0.f;
    for (int c = 0; c < C; ++c){ float v = X[((size_t)b * C + c) * T + t]; s += v * v; }
    float inv = rsqrtf(s);
    for (int c = 0; c < C; ++c) X[((size_t)b * C + c) * T + t] *= inv;
}

// ---------------- VQ: argmin distance to normalized codebook, accumulate diff -------
__global__ void k_quantize(const float* __restrict__ X, const float* __restrict__ W,
                           float* __restrict__ idx_out, float* __restrict__ diff,
                           int T, float inv_count){
    __shared__ float nrm[128];
    int tid = threadIdx.x;
    {
        float s = 0.f; const float* wr = W + tid * 80;
        for (int d = 0; d < 80; ++d) s += wr[d] * wr[d];
        nrm[tid] = sqrtf(s);
    }
    __syncthreads();
    long long token = (long long)blockIdx.x * 128 + tid;
    if (token >= (long long)8 * T) return;
    int b = (int)(token / T), t = (int)(token % T);
    float xv[80]; float xx = 0.f;
    for (int d = 0; d < 80; ++d){ float v = X[((size_t)b * 80 + d) * T + t]; xv[d] = v; xx += v * v; }
    int best = 0; float bd = 3.4e38f;
    for (int j = 0; j < 128; ++j){
        const float* wr = W + j * 80;
        float dot = 0.f;
        for (int d = 0; d < 80; ++d) dot += xv[d] * wr[d];
        float dist = xx - 2.f * dot / nrm[j] + 1.f;
        if (dist < bd){ bd = dist; best = j; }
    }
    float nb = nrm[best];
    float sc = (nb > 1.f) ? 1.f / (nb + 1e-7f) : 1.f;
    float ds = 0.f; const float* wb = W + best * 80;
    for (int d = 0; d < 80; ++d){ float r = wb[d] * sc - xv[d]; ds += r * r; }
    atomicAdd(diff, ds * inv_count);
    idx_out[token] = (float)best;
}

// ---------------- misc elementwise ----------------
__global__ void k_up2(const float* __restrict__ X, float* __restrict__ Y, long long BC, int T){
    long long i = (long long)blockIdx.x * blockDim.x + threadIdx.x;
    long long tot = BC * (long long)(2 * T);
    if (i >= tot) return;
    long long bc = i / (2 * T); int t = (int)(i % (2 * T));
    Y[i] = X[bc * T + (t >> 1)];
}
__global__ void k_add_tr(const float* __restrict__ F, const float* __restrict__ R,
                         float* __restrict__ Y, int T){
    long long i = (long long)blockIdx.x * blockDim.x + threadIdx.x;
    long long tot = (long long)8 * 512 * T;
    if (i >= tot) return;
    int b = (int)(i / ((long long)512 * T));
    long long rem = i - (long long)b * 512 * T;
    int c = (int)(rem / T), t = (int)(rem % T);
    size_t src = ((size_t)b * T + t) * 512 + c;
    Y[i] = F[src] + R[src];
}
__global__ void k_add3(float* __restrict__ Y, const float* __restrict__ A2,
                       const float* __restrict__ A3, long long n){
    long long i = (long long)blockIdx.x * blockDim.x + threadIdx.x;
    if (i >= n) return;
    Y[i] = Y[i] + A2[i] + A3[i];
}

// ---------------- host helpers ----------------
static void h_im2col(hipStream_t s, const float* X, unsigned short* A, int Cin, int ci_off,
                     int Ctot, int Tin, int Tout, int KW, int stride, int pad,
                     int reflect, int flip, int Kp){
    long long M = 8LL * Tout;
    long long tot = M * Kp;
    k_im2col<<<(int)CDIV(tot, 256), 256, 0, s>>>(X, A, Cin, ci_off, Ctot, Tin, Tout,
                                                 KW, stride, pad, reflect, flip, Kp, M, Cin * KW);
}
static void h_pack(hipStream_t s, const float* W, unsigned short* P, int N, int K, int ldw, int Kp){
    int NT = CDIV(N, 16), KT = Kp / 32;
    long long tot = (long long)NT * KT * 32;
    k_pack<<<(int)CDIV(tot, 256), 256, 0, s>>>(W, P, N, K, ldw, KT, NT);
}
static void h_gemm(hipStream_t s, const unsigned short* A, const unsigned short* P,
                   const float* bias, float* out, int M, int N, int Kp, int mode,
                   int T, int Cout, int n_off){
    int NT = CDIV(N, 16), Mt = M / 16;
    int tiles = Mt * NT;
    k_gemm<<<CDIV(tiles, 8), 256, 0, s>>>(A, P, bias, out, M, N, Kp, NT, tiles, mode, T, Cout, n_off);
}

extern "C" void kernel_launch(void* const* d_in, const int* in_sizes, int n_in,
                              void* d_out, int out_size, void* d_ws, size_t ws_size,
                              hipStream_t stream){
    const int B = 8, CIN = 80, CH = 512, T0 = 1024;
    (void)in_sizes; (void)n_in; (void)out_size; (void)ws_size;
    auto IN = [&](int i){ return (const float*)d_in[i]; };

    char* wsb = (char*)d_ws; size_t off = 0;
    auto alloc = [&](size_t bytes)->char*{
        off = (off + 255) & ~(size_t)255; char* p = wsb + off; off += bytes; return p;
    };
    unsigned short* bufA = (unsigned short*)alloc((size_t)8192 * 1536 * 2);
    float* xg_f  = (float*)alloc((size_t)8192 * 1536 * 4);
    float* xg_b  = (float*)alloc((size_t)8192 * 1536 * 4);
    float* gru_f = (float*)alloc((size_t)8192 * 512 * 4);
    float* gru_b = (float*)alloc((size_t)8192 * 512 * 4);
    float* ybuf  = (float*)alloc((size_t)B * CH * 1024 * 4);
    float* rbuf  = (float*)alloc((size_t)B * CH * 1024 * 4);
    float* cbuf  = (float*)alloc((size_t)B * CH * 1024 * 4);
    float* xb0   = (float*)alloc((size_t)B * CH * 1024 * 4);
    float* xb1   = (float*)alloc((size_t)B * CH * 1024 * 4);
    float* xh0   = (float*)alloc((size_t)B * CIN * 1024 * 4);
    float* xh1   = (float*)alloc((size_t)B * CIN * 1024 * 4);
    float* code  = (float*)alloc((size_t)B * CIN * 1024 * 4);
    float* enc_out[3] = { (float*)alloc((size_t)B * CIN * 512 * 4),
                          (float*)alloc((size_t)B * CIN * 256 * 4),
                          (float*)alloc((size_t)B * CIN * 128 * 4) };
    unsigned short* wp = (unsigned short*)alloc((size_t)2 * 1024 * 1024);

    float* outp = (float*)d_out;
    const size_t OFF_O[3] = {0, 163840, 491520};
    const size_t OFF_DIFF = 1146880;
    const size_t OFF_I[3] = {1146881, 1150977, 1153025};
    hipMemsetAsync(outp + OFF_DIFF, 0, sizeof(float), stream);

    // ================= Encoder =================
    const float* cur = IN(0);
    int Tin = T0;
    for (int s = 0; s < 3; ++s){
        int Tout = Tin / 2;
        int bse = 1 + s * 6;
        const float *w1 = IN(bse), *b1 = IN(bse+1), *w2 = IN(bse+2), *b2 = IN(bse+3);
        const float *gnw = IN(bse+4), *gnb = IN(bse+5);
        // conv1 80->512 k4 s2 reflect pad1, leaky
        h_im2col(stream, cur, bufA, CIN, 0, CIN, Tin, Tout, 4, 2, 1, 1, 0, 320);
        h_pack(stream, w1, wp, CH, 320, 320, 320);
        h_gemm(stream, bufA, wp, b1, ybuf, B*Tout, CH, 320, 1|2, Tout, CH, 0);
        // conv2 512->80 k3 reflect pad1
        h_im2col(stream, ybuf, bufA, CH, 0, CH, Tout, Tout, 3, 1, 1, 1, 0, 1536);
        h_pack(stream, w2, wp, CIN, 1536, 1536, 1536);
        h_gemm(stream, bufA, wp, b2, enc_out[s], B*Tout, CIN, 1536, 2, Tout, CIN, 0);
        k_groupnorm<<<B*2, 256, 0, stream>>>(enc_out[s], enc_out[s], gnw, gnb, CIN, Tout, 2, 0);
        k_l2norm<<<CDIV(B*Tout, 256), 256, 0, stream>>>(enc_out[s], CIN, Tout);
        k_quantize<<<CDIV(B*Tout, 128), 128, 0, stream>>>(enc_out[s], IN(19 + s),
            outp + OFF_I[s], outp + OFF_DIFF, Tout, 1.f / (float)(B * Tout * CIN));
        cur = enc_out[s]; Tin = Tout;
    }

    // ================= Decoder =================
    float *xb_cur = xb0, *xb_alt = xb1, *xh_cur = xh0, *xh_alt = xh1;
    for (int i = 0; i < 3; ++i){
        int Tc = 256 << i;
        int bb = 22 + i * 18;
        const float *c1w = IN(bb+0),  *c1b = IN(bb+1),  *gn1w = IN(bb+2), *gn1b = IN(bb+3);
        const float *wihf= IN(bb+4),  *whhf= IN(bb+5),  *bihf = IN(bb+6), *bhhf = IN(bb+7);
        const float *wihb= IN(bb+8),  *whhb= IN(bb+9),  *bihb = IN(bb+10),*bhhb = IN(bb+11);
        const float *rcw = IN(bb+12), *rcb = IN(bb+13), *rgnw = IN(bb+14),*rgnb = IN(bb+15);
        const float *c2w = IN(bb+16), *c2b = IN(bb+17);
        const float *hw  = IN(76 + i*2), *hb2 = IN(77 + i*2);
        const float *ibw = IN(82 + i*2), *ibb = IN(83 + i*2);
        const float *h0  = IN(88 + i);

        // code = up2(q[i]) ; q[i] = enc_out[2-i]
        {
            long long tot = (long long)B * CIN * Tc;
            k_up2<<<(int)CDIV(tot,256), 256, 0, stream>>>(enc_out[2 - i], code, (long long)B*CIN, Tc/2);
        }
        if (i == 0){
            hipMemsetAsync(xb_cur, 0, (size_t)B*CH*Tc*4, stream);
            hipMemsetAsync(xh_cur, 0, (size_t)B*CIN*Tc*4, stream);
        } else {
            long long t1 = (long long)B*CH*Tc, t2 = (long long)B*CIN*Tc;
            k_up2<<<(int)CDIV(t1,256), 256, 0, stream>>>(xb_cur, xb_alt, (long long)B*CH, Tc/2);
            k_up2<<<(int)CDIV(t2,256), 256, 0, stream>>>(xh_cur, xh_alt, (long long)B*CIN, Tc/2);
            float* t; t = xb_cur; xb_cur = xb_alt; xb_alt = t;
                      t = xh_cur; xh_cur = xh_alt; xh_alt = t;
        }

        // ---- gblock ----
        // c1: 80->512 k3 zero-pad1 ; then GN(4)+leaky
        h_im2col(stream, code, bufA, CIN, 0, CIN, Tc, Tc, 3, 1, 1, 0, 0, 256);
        h_pack(stream, c1w, wp, CH, 240, 240, 256);
        h_gemm(stream, bufA, wp, c1b, ybuf, B*Tc, CH, 256, 2, Tc, CH, 0);
        k_groupnorm<<<B*4, 256, 0, stream>>>(ybuf, ybuf, gn1w, gn1b, CH, Tc, 4, 1);

        // rcblock: bigru input projections
        h_im2col(stream, ybuf, bufA, CH, 0, CH, Tc, Tc, 1, 1, 0, 0, 0, 512);
        h_pack(stream, wihf, wp, 1536, 512, 512, 512);
        h_gemm(stream, bufA, wp, bihf, xg_f, B*Tc, 1536, 512, 0, 1, 0, 0);
        h_im2col(stream, ybuf, bufA, CH, 0, CH, Tc, Tc, 1, 1, 0, 0, 1, 512);
        h_pack(stream, wihb, wp, 1536, 512, 512, 512);
        h_gemm(stream, bufA, wp, bihb, xg_b, B*Tc, 1536, 512, 0, 1, 0, 0);
        // recurrent passes (persistent WMMA kernels)
        h_pack(stream, whhf, wp, 1536, 512, 512, 512);
        k_gru<<<1, 512, 0, stream>>>(xg_f, wp, bhhf, h0, gru_f, Tc, 0);
        h_pack(stream, whhb, wp, 1536, 512, 512, 512);
        k_gru<<<1, 512, 0, stream>>>(xg_b, wp, bhhb, h0 + (size_t)B*CH, gru_b, Tc, 1);
        {
            long long tot = (long long)B*CH*Tc;
            k_add_tr<<<(int)CDIV(tot,256), 256, 0, stream>>>(gru_f, gru_b, rbuf, Tc);
        }
        // grouped conv (4 groups of 128) on rbuf -> cbuf, then GN(4)+leaky
        for (int g = 0; g < 4; ++g){
            h_im2col(stream, rbuf, bufA, 128, g*128, CH, Tc, Tc, 3, 1, 1, 0, 0, 384);
            h_pack(stream, rcw + (size_t)g*128*384, wp, 128, 384, 384, 384);
            h_gemm(stream, bufA, wp, rcb + g*128, cbuf, B*Tc, 128, 384, 2, Tc, CH, g*128);
        }
        k_groupnorm<<<B*4, 256, 0, stream>>>(cbuf, cbuf, rgnw, rgnb, CH, Tc, 4, 1);
        // y = y + r + c
        {
            long long tot = (long long)B*CH*Tc;
            k_add3<<<(int)CDIV(tot,256), 256, 0, stream>>>(ybuf, rbuf, cbuf, tot);
        }
        // c2: 512->512 k3 zero-pad1, accumulate into x_body
        h_im2col(stream, ybuf, bufA, CH, 0, CH, Tc, Tc, 3, 1, 1, 0, 0, 1536);
        h_pack(stream, c2w, wp, CH, 1536, 1536, 1536);
        h_gemm(stream, bufA, wp, c2b, xb_cur, B*Tc, CH, 1536, 2|4, Tc, CH, 0);

        // inblock GN
        k_groupnorm<<<B*4, 256, 0, stream>>>(xb_cur, xb_cur, ibw, ibb, CH, Tc, 4, 0);
        // head: 512->80 k3 reflect pad1, accumulate into x_head
        h_im2col(stream, xb_cur, bufA, CH, 0, CH, Tc, Tc, 3, 1, 1, 1, 0, 1536);
        h_pack(stream, hw, wp, CIN, 1536, 1536, 1536);
        h_gemm(stream, bufA, wp, hb2, xh_cur, B*Tc, CIN, 1536, 2|4, Tc, CIN, 0);

        hipMemcpyAsync(outp + OFF_O[i], xh_cur, (size_t)B*CIN*Tc*4,
                       hipMemcpyDeviceToDevice, stream);
    }
}
